// ScaledDotProductAttention_4312147165222
// MI455X (gfx1250) — matrix operand order repaired and checked by reading, not compiled, not run on hardware
//
#include <hip/hip_runtime.h>
#include <hip/hip_bf16.h>
#include <math.h>

// ---- CDNA5 WMMA vector types ----
typedef __attribute__((ext_vector_type(16))) _Float16 v16h;
typedef __attribute__((ext_vector_type(8)))  float    v8f;
typedef __attribute__((ext_vector_type(4)))  float    f4;

#define S_LEN   2048
#define DHEAD   64
#define NHEADS  16
#define NBATCH  2
#define COLS_PER_WAVE 256
#define NEG_INF_F (-3.402823466e38f)

// Block: 256 threads (8 waves of 32). Each block owns a 16-row query tile of
// one (b,h). Each wave owns a 256-column slab of the 2048 key columns.
__global__ __launch_bounds__(256) void attn_fwd_kernel(
    const float* __restrict__ Q, const float* __restrict__ K,
    const float* __restrict__ V, const int* __restrict__ mask,
    float* __restrict__ out, float* __restrict__ attnw)
{
    extern __shared__ float smem[];
    float* sc     = smem;                    // [16][2048] fp32 scores / exp
    float* rowinv = sc + 16 * S_LEN;         // [16] 1/rowsum
    float* opart  = rowinv + 16;             // [8][16][64] per-wave partial O

    const int tid  = threadIdx.x;
    const int lane = tid & 31;
    const int wave = tid >> 5;
    const int l16  = lane & 15;
    const bool hi  = lane >= 16;

    const int qtiles = S_LEN / 16;
    const int bh = blockIdx.x / qtiles;       // fused batch*head
    const int qt = blockIdx.x % qtiles;
    const int q0 = qt * 16;
    const int b  = bh / NHEADS;

    const size_t headOff = (size_t)bh * S_LEN * DHEAD;
    const float* Qb = Q + headOff;
    const float* Kb = K + headOff;
    const float* Vb = V + headOff;
    const int*   Mb = mask + (size_t)b * S_LEN * S_LEN + (size_t)q0 * S_LEN;

    // ---------------- Q A-fragments (16x32 f16) for the two D-chunks -------
    // A layout: lane L -> row M=L&15; halves 0..7 -> K = base+0..7,
    //           halves 8..15 -> K = base+16..23, base = chunk*32 + (hi?8:0).
    v16h aQ[2];
    {
        const float* qrow = Qb + (size_t)(q0 + l16) * DHEAD;
        #pragma unroll
        for (int c = 0; c < 2; ++c) {
            const int baseD = c * 32 + (hi ? 8 : 0);
            f4 p0 = *reinterpret_cast<const f4*>(qrow + baseD + 0);
            f4 p1 = *reinterpret_cast<const f4*>(qrow + baseD + 4);
            f4 p2 = *reinterpret_cast<const f4*>(qrow + baseD + 16);
            f4 p3 = *reinterpret_cast<const f4*>(qrow + baseD + 20);
            v16h a;
            #pragma unroll
            for (int i = 0; i < 4; ++i) {
                a[i]      = (_Float16)p0[i];
                a[4 + i]  = (_Float16)p1[i];
                a[8 + i]  = (_Float16)p2[i];
                a[12 + i] = (_Float16)p3[i];
            }
            aQ[c] = a;
        }
    }

    // ---------------- Phase 1: S = (Q K^T) * 1/sqrt(D), masked, to LDS -----
    const int colW = wave * COLS_PER_WAVE;
    for (int nt = 0; nt < 16; ++nt) {
        const int n = colW + nt * 16 + l16;   // absolute key column this lane owns
        v8f acc = {};
        #pragma unroll
        for (int c = 0; c < 2; ++c) {
            // B layout: lane -> col N=l16; halves 0..7 -> K = base+0..7,
            // halves 8..15 -> K = base+16..23, base = chunk*32 + (hi?8:0).
            const float* krow = Kb + (size_t)n * DHEAD + c * 32 + (hi ? 8 : 0);
            f4 k0 = *reinterpret_cast<const f4*>(krow + 0);
            f4 k1 = *reinterpret_cast<const f4*>(krow + 4);
            f4 k2 = *reinterpret_cast<const f4*>(krow + 16);
            f4 k3 = *reinterpret_cast<const f4*>(krow + 20);
            v16h bk;
            #pragma unroll
            for (int i = 0; i < 4; ++i) {
                bk[i]      = (_Float16)k0[i];
                bk[4 + i]  = (_Float16)k1[i];
                bk[8 + i]  = (_Float16)k2[i];
                bk[12 + i] = (_Float16)k3[i];
            }
            acc = __builtin_amdgcn_wmma_f32_16x16x32_f16(
                false, aQ[c], false, bk, (short)0, acc, false, false);
        }
        // C/D layout: VGPR r -> row M = r + (hi?8:0); lane -> col N=l16.
        #pragma unroll
        for (int r = 0; r < 8; ++r) {
            const int row = r + (hi ? 8 : 0);
            float s = acc[r] * 0.125f;               // 1/sqrt(64)
            if (Mb[(size_t)row * S_LEN + n] == 0) s = NEG_INF_F;
            sc[row * S_LEN + n] = s;
        }
    }
    __syncthreads();

    // ---------------- Phase 2: exact fp32 softmax + attn_weights write -----
    {
        const int row = tid >> 4;            // 16 threads per row (one half-wave)
        const int sub = tid & 15;
        float* srow = sc + row * S_LEN;

        float m = NEG_INF_F;
        for (int i = sub; i < S_LEN; i += 16) m = fmaxf(m, srow[i]);
        #pragma unroll
        for (int off = 8; off >= 1; off >>= 1) m = fmaxf(m, __shfl_xor(m, off, 16));

        float sum = 0.f;
        for (int i = sub; i < S_LEN; i += 16) {
            float e = __expf(srow[i] - m);   // masked -> exp(-3.4e38) == 0
            srow[i] = e;
            sum += e;
        }
        #pragma unroll
        for (int off = 8; off >= 1; off >>= 1) sum += __shfl_xor(sum, off, 16);

        const float inv = 1.0f / sum;
        if (sub == 0) rowinv[row] = inv;

        float* awrow = attnw + ((size_t)bh * S_LEN + q0 + row) * S_LEN;
        for (int i = 0; i < S_LEN / 16; ++i) {
            const int col = i * 16 + sub;    // consecutive lanes -> consecutive cols
            awrow[col] = srow[col] * inv;
        }
    }
    __syncthreads();

    // ---------------- Phase 3: O = P V (each wave contracts its 256 cols) --
    v8f accO[4] = {};
    for (int kt = 0; kt < 8; ++kt) {
        const int kb = colW + kt * 32;
        // A fragment from LDS exp-scores, normalized + cvt to f16
        const float invm = rowinv[l16];
        const float* prow = sc + l16 * S_LEN + kb + (hi ? 8 : 0);
        v16h ap;
        #pragma unroll
        for (int h = 0; h < 8; ++h) ap[h]     = (_Float16)(prow[h] * invm);
        #pragma unroll
        for (int h = 0; h < 8; ++h) ap[8 + h] = (_Float16)(prow[16 + h] * invm);

        #pragma unroll
        for (int nt = 0; nt < 4; ++nt) {
            const int n    = nt * 16 + l16;
            const int kcol = kb + (hi ? 8 : 0);
            const float* vcol = Vb + (size_t)kcol * DHEAD + n;
            v16h bv;
            #pragma unroll
            for (int h = 0; h < 16; ++h)
                bv[h] = (_Float16)vcol[(size_t)(h < 8 ? h : h + 8) * DHEAD]; // coalesced across lanes
            accO[nt] = __builtin_amdgcn_wmma_f32_16x16x32_f16(
                false, ap, false, bv, (short)0, accO[nt], false, false);
        }
    }
    // spill per-wave partial 16x64 tiles
    #pragma unroll
    for (int nt = 0; nt < 4; ++nt) {
        #pragma unroll
        for (int r = 0; r < 8; ++r) {
            const int row = r + (hi ? 8 : 0);
            opart[(wave * 16 + row) * DHEAD + nt * 16 + l16] = accO[nt][r];
        }
    }
    __syncthreads();

    // ---------------- Phase 4: 8-wave reduction + output store -------------
    for (int e = tid; e < 16 * DHEAD; e += 256) {
        float s = 0.f;
        #pragma unroll
        for (int w = 0; w < 8; ++w) s += opart[w * 16 * DHEAD + e];
        out[((size_t)bh * S_LEN + q0 + (e / DHEAD)) * DHEAD + (e % DHEAD)] = s;
    }
}

extern "C" void kernel_launch(void* const* d_in, const int* in_sizes, int n_in,
                              void* d_out, int out_size, void* d_ws, size_t ws_size,
                              hipStream_t stream) {
    const float* Q    = (const float*)d_in[0];
    const float* K    = (const float*)d_in[1];
    const float* V    = (const float*)d_in[2];
    const int*   mask = (const int*)d_in[3];

    float* out   = (float*)d_out;                                  // [B,H,S,DV]
    float* attnw = out + (size_t)NBATCH * NHEADS * S_LEN * DHEAD;  // [B,H,S,S]

    const size_t smemBytes =
        (size_t)(16 * S_LEN + 16 + 8 * 16 * DHEAD) * sizeof(float); // ~160 KB < 320 KB WGP LDS

    dim3 grid(NBATCH * NHEADS * (S_LEN / 16));
    attn_fwd_kernel<<<grid, 256, smemBytes, stream>>>(Q, K, V, mask, out, attnw);
}
